// ReasoningFunction_62740882260555
// MI455X (gfx1250) — matrix operand order repaired and checked by reading, not compiled, not run on hardware
//
#include <hip/hip_runtime.h>
#include <math.h>

#define DD 1024
#define NN 4096
#define KTOP 16
#define EE 3072

typedef __attribute__((ext_vector_type(16))) __bf16 v16bf;
typedef __attribute__((ext_vector_type(8)))  float  v8f;

union Frag { uint4 q[2]; v16bf v; };

__device__ __forceinline__ unsigned short f2bf(float f) {
  union { float f; unsigned u; } v; v.f = f;
  unsigned r = v.u + 0x7FFFu + ((v.u >> 16) & 1u);   // round-to-nearest-even
  return (unsigned short)(r >> 16);
}
__device__ __forceinline__ unsigned pack2(float a, float b) {
  return (unsigned)f2bf(a) | ((unsigned)f2bf(b) << 16);
}
__device__ __forceinline__ v8f v8f_zero() {
  v8f z;
#pragma unroll
  for (int i = 0; i < 8; ++i) z[i] = 0.f;
  return z;
}

// Async global->LDS copy of 16 bytes (CDNA5 ASYNCcnt path).
__device__ __forceinline__ void async_ld16(unsigned ldsoff, const void* gaddr) {
  asm volatile("global_load_async_to_lds_b128 %0, %1, off"
               :: "v"(ldsoff), "v"(gaddr) : "memory");
}
__device__ __forceinline__ void wait_async0() {
  asm volatile("s_wait_asynccnt 0x0" ::: "memory");
}

// ---------------------------------------------------------------------------
// Prep: W_logic -> bf16 transposed [D][3D]; W_gate -> bf16 [D][2D] with the
// two arg-summary blocks pre-summed; token -> bf16 copy (row-major).
// ---------------------------------------------------------------------------
__global__ void prep_kernel(const float* __restrict__ Wl, const float* __restrict__ Wg,
                            const float* __restrict__ token,
                            unsigned short* __restrict__ wlt, unsigned short* __restrict__ wgt,
                            unsigned short* __restrict__ tokbf) {
  const long long T1 = (long long)DD * EE;
  const long long T2 = (long long)DD * 2048;
  const long long T3 = (long long)NN * DD;
  long long idx = (long long)blockIdx.x * blockDim.x + threadIdx.x;
  if (idx < T1) {
    int d = (int)(idx / EE), e = (int)(idx % EE);
    wlt[idx] = f2bf(Wl[(size_t)e * DD + d]);
  } else if (idx < T1 + T2) {
    long long j = idx - T1;
    int d = (int)(j / 2048), e = (int)(j % 2048);
    float v;
    if (e < DD) v = Wg[(size_t)e * DD + d];
    else {
      int e2 = e - DD;
      v = Wg[(size_t)(DD + e2) * DD + d] + Wg[(size_t)(2 * DD + e2) * DD + d];
    }
    wgt[j] = f2bf(v);
  } else if (idx < T1 + T2 + T3) {
    long long j = idx - T1 - T2;
    tokbf[j] = f2bf(token[j]);
  }
}

// ---------------------------------------------------------------------------
// Logic GEMM + gelu + layernorm + softmax-weighted sum -> arg_summary[N,D]
// One block per token n. 256 threads = 8 wave32; wave w owns cols [128w,128w+128).
// ---------------------------------------------------------------------------
__global__ __launch_bounds__(256) void logic_kernel(
    const float* __restrict__ token, const float* __restrict__ routed,
    const float* __restrict__ delta, const float* __restrict__ sim,
    const unsigned short* __restrict__ Wt, const float* __restrict__ bias,
    const float* __restrict__ gamma, const float* __restrict__ beta,
    float* __restrict__ outArg, unsigned short* __restrict__ argbf) {
  __shared__ __attribute__((aligned(16))) unsigned short Abuf[2][16][520];
  __shared__ float red_s[16], red_q[16], wsm[16], simb[16];

  const int tid  = threadIdx.x;
  const int n    = blockIdx.x;
  const int lane = tid & 31;
  const int wave = tid >> 5;
  const bool lo  = lane < 16;
  const int c16  = lane & 15;
  const int wcol = wave * 128;
  const int r    = tid >> 4;          // A-load row 0..15
  const int cc   = (tid & 15) * 32;   // A-load col chunk base within stage

  v8f acc[8];
#pragma unroll
  for (int t = 0; t < 8; ++t) acc[t] = v8f_zero();

  float4 pre[8];
  auto loadRegs = [&](int s) {
    const int seg = s >> 1, off = (s & 1) * 512;
    const float* src;
    if (seg == 0)      src = token  + (size_t)n * DD + off + cc;
    else if (seg == 1) src = routed + ((size_t)n * KTOP + r) * DD + off + cc;
    else               src = delta  + ((size_t)n * KTOP + r) * DD + off + cc;
#pragma unroll
    for (int j = 0; j < 8; ++j) pre[j] = ((const float4*)src)[j];
  };
  auto storeRegs = [&](int buf) {
#pragma unroll
    for (int j = 0; j < 8; ++j) {
      uint2 u; u.x = pack2(pre[j].x, pre[j].y); u.y = pack2(pre[j].z, pre[j].w);
      *(uint2*)&Abuf[buf][r][cc + j * 4] = u;
    }
  };

  // B-fragment base pointers: lane holds column (wcol+16t+c16), K-contiguous.
  const unsigned short* wb[8];
#pragma unroll
  for (int t = 0; t < 8; ++t)
    wb[t] = Wt + (size_t)(wcol + t * 16 + c16) * EE + (lo ? 0 : 8);

  loadRegs(0); storeRegs(0); __syncthreads();

  for (int s = 0; s < 6; ++s) {
    const int cur = s & 1;
    if (s < 5) loadRegs(s + 1);                 // overlap global latency with WMMA
    const unsigned short* arow = &Abuf[cur][c16][0];
    const int aoff = lo ? 0 : 8;                // ISA 16-bit A layout: lanes>=16 hold K {8..15,24..31}
#pragma unroll 4
    for (int kk = 0; kk < 16; ++kk) {
      Frag af;
      af.q[0] = *(const uint4*)(arow + kk * 32 + aoff);
      af.q[1] = *(const uint4*)(arow + kk * 32 + aoff + 16);
#pragma unroll
      for (int t = 0; t < 8; ++t) {
        Frag bf_;
        const unsigned short* bp = wb[t] + (size_t)s * 512 + kk * 32;
        bf_.q[0] = *(const uint4*)bp;
        bf_.q[1] = *(const uint4*)(bp + 16);
        acc[t] = __builtin_amdgcn_wmma_f32_16x16x32_bf16(
            false, af.v, false, bf_.v, (short)0, acc[t], false, false);
      }
    }
    if (s < 5) { storeRegs(cur ^ 1); __syncthreads(); }
  }

  // ---- epilogue: bias + exact gelu, layernorm stats, softmax-weighted sum ----
  if (tid < 16) { red_s[tid] = 0.f; red_q[tid] = 0.f; simb[tid] = sim[(size_t)n * KTOP + tid]; }
  __syncthreads();

  float g[8][8];
  float ps[8], pq[8];
#pragma unroll
  for (int v = 0; v < 8; ++v) { ps[v] = 0.f; pq[v] = 0.f; }
#pragma unroll
  for (int t = 0; t < 8; ++t) {
    const int col = wcol + t * 16 + c16;
    const float b = bias[col];
#pragma unroll
    for (int v = 0; v < 8; ++v) {
      float x = acc[t][v] + b;
      float gg = 0.5f * x * (1.0f + erff(x * 0.7071067811865475f));
      g[t][v] = gg;
      ps[v] += gg; pq[v] += gg * gg;
    }
  }
#pragma unroll
  for (int m = 1; m < 16; m <<= 1) {
#pragma unroll
    for (int v = 0; v < 8; ++v) {
      ps[v] += __shfl_xor(ps[v], m, 32);
      pq[v] += __shfl_xor(pq[v], m, 32);
    }
  }
  if (c16 == 0) {
    const int rb0 = lo ? 0 : 8;
#pragma unroll
    for (int v = 0; v < 8; ++v) {
      atomicAdd(&red_s[rb0 + v], ps[v]);
      atomicAdd(&red_q[rb0 + v], pq[v]);
    }
  }
  __syncthreads();
  if (tid == 0) {
    float mx = simb[0];
    for (int i = 1; i < 16; ++i) mx = fmaxf(mx, simb[i]);
    float sum = 0.f;
    for (int i = 0; i < 16; ++i) { float e = expf(simb[i] - mx); wsm[i] = e; sum += e; }
    float inv = 1.0f / sum;
    for (int i = 0; i < 16; ++i) wsm[i] *= inv;
  }
  if (tid < 16) {
    float mu  = red_s[tid] * (1.0f / DD);
    float var = red_q[tid] * (1.0f / DD) - mu * mu;
    red_s[tid] = mu;
    red_q[tid] = rsqrtf(var + 1e-5f);
  }
  __syncthreads();

  const int rb = lo ? 0 : 8;
#pragma unroll
  for (int t = 0; t < 8; ++t) {
    const int col = wcol + t * 16 + c16;
    const float ga = gamma[col], be = beta[col];
    float part = 0.f;
#pragma unroll
    for (int v = 0; v < 8; ++v) {
      const int rr = rb + v;
      float f = (g[t][v] - red_s[rr]) * red_q[rr] * ga + be;
      part += wsm[rr] * f;
    }
    part += __shfl_xor(part, 16, 32);   // lanes l / l+16 hold same col, complementary rows
    outArg[(size_t)n * DD + col] = part;
    argbf[(size_t)n * DD + col] = f2bf(part);   // bf16 copy for async-staged gate GEMM
  }
}

// ---------------------------------------------------------------------------
// Gate: sigmoid([token, arg] @ W_eff + b). 16 tokens per block, K=2048.
// A operand is already bf16 in memory -> stage tiles into LDS with
// global_load_async_to_lds_b128 (ASYNCcnt double buffering, no VALU packing).
// ---------------------------------------------------------------------------
__global__ __launch_bounds__(256) void gate_kernel(
    const unsigned short* __restrict__ tokbf, const unsigned short* __restrict__ argbf,
    const unsigned short* __restrict__ Wt, const float* __restrict__ bias,
    float* __restrict__ outGate) {
  __shared__ __attribute__((aligned(16))) unsigned short Abuf[2][16][520];

  const int tid  = threadIdx.x;
  const int n0   = blockIdx.x * 16;
  const int lane = tid & 31;
  const int wave = tid >> 5;
  const bool lo  = lane < 16;
  const int c16  = lane & 15;
  const int wcol = wave * 128;
  const int r    = tid >> 4;          // A-stage row 0..15
  const int cc   = (tid & 15) * 32;   // A-stage col chunk base (elements)

  v8f acc[8];
#pragma unroll
  for (int t = 0; t < 8; ++t) acc[t] = v8f_zero();

  // Issue async copies for one 16x512 bf16 stage: 4 x 16B chunks per thread.
  auto issueStage = [&](int s, int buf) {
    const unsigned short* src = ((s < 2) ? tokbf : argbf)
                              + (size_t)(n0 + r) * DD + (s & 1) * 512 + cc;
#pragma unroll
    for (int j = 0; j < 4; ++j) {
      unsigned ldsoff = (unsigned)(size_t)&Abuf[buf][r][cc + j * 8];
      async_ld16(ldsoff, src + j * 8);
    }
  };

  const unsigned short* wb[8];
#pragma unroll
  for (int t = 0; t < 8; ++t)
    wb[t] = Wt + (size_t)(wcol + t * 16 + c16) * 2048 + (lo ? 0 : 8);

  issueStage(0, 0);
  wait_async0();
  __syncthreads();

  for (int s = 0; s < 4; ++s) {
    const int cur = s & 1;
    if (s < 3) issueStage(s + 1, cur ^ 1);      // async copy overlaps WMMA compute
    const unsigned short* arow = &Abuf[cur][c16][0];
    const int aoff = lo ? 0 : 8;
#pragma unroll 4
    for (int kk = 0; kk < 16; ++kk) {
      Frag af;
      af.q[0] = *(const uint4*)(arow + kk * 32 + aoff);
      af.q[1] = *(const uint4*)(arow + kk * 32 + aoff + 16);
#pragma unroll
      for (int t = 0; t < 8; ++t) {
        Frag bf_;
        const unsigned short* bp = wb[t] + (size_t)s * 512 + kk * 32;
        bf_.q[0] = *(const uint4*)bp;
        bf_.q[1] = *(const uint4*)(bp + 16);
        acc[t] = __builtin_amdgcn_wmma_f32_16x16x32_bf16(
            false, af.v, false, bf_.v, (short)0, acc[t], false, false);
      }
    }
    if (s < 3) { wait_async0(); __syncthreads(); }
  }

  const int rb = lo ? 0 : 8;
#pragma unroll
  for (int t = 0; t < 8; ++t) {
    const int col = wcol + t * 16 + c16;
    const float b = bias[col];
#pragma unroll
    for (int v = 0; v < 8; ++v) {
      float x = acc[t][v] + b;
      float sg = 1.0f / (1.0f + expf(-x));
      outGate[(size_t)(n0 + rb + v) * DD + col] = sg;
    }
  }
}

// ---------------------------------------------------------------------------
extern "C" void kernel_launch(void* const* d_in, const int* in_sizes, int n_in,
                              void* d_out, int out_size, void* d_ws, size_t ws_size,
                              hipStream_t stream) {
  const float* token = (const float*)d_in[0];
  const float* routed = (const float*)d_in[1];
  const float* sim   = (const float*)d_in[2];
  const float* delta = (const float*)d_in[3];
  const float* Wl    = (const float*)d_in[4];
  const float* bl    = (const float*)d_in[5];
  const float* gamma = (const float*)d_in[6];
  const float* beta  = (const float*)d_in[7];
  const float* Wg    = (const float*)d_in[8];
  const float* bg    = (const float*)d_in[9];

  float* outArg  = (float*)d_out;
  float* outGate = outArg + (size_t)NN * DD;

  unsigned short* wlt   = (unsigned short*)d_ws;               // bf16 [D][3D]
  unsigned short* wgt   = wlt + (size_t)DD * EE;               // bf16 [D][2D]
  unsigned short* tokbf = wgt + (size_t)DD * 2048;             // bf16 [N][D]
  unsigned short* argbf = tokbf + (size_t)NN * DD;             // bf16 [N][D]

  long long total = (long long)DD * EE + (long long)DD * 2048 + (long long)NN * DD;
  int pblocks = (int)((total + 255) / 256);
  prep_kernel<<<pblocks, 256, 0, stream>>>(Wl, Wg, token, wlt, wgt, tokbf);
  logic_kernel<<<NN, 256, 0, stream>>>(token, routed, delta, sim, wlt, bl, gamma, beta,
                                       outArg, argbf);
  gate_kernel<<<NN / 16, 256, 0, stream>>>(tokbf, argbf, wgt, bg, outGate);
}
